// TMM_2_85925115724265
// MI455X (gfx1250) — matrix operand order repaired and checked by reading, not compiled, not run on hardware
//
#include <hip/hip_runtime.h>
#include <hip/hip_bf16.h>
#include <math.h>

// ---------------- problem constants ----------------
#define T_STEPS 16384
#define XDIM 128
#define ZDIM 64
#define HDIM 128

typedef __attribute__((ext_vector_type(16))) _Float16 v16h;
typedef __attribute__((ext_vector_type(8)))  _Float16 v8h;
typedef __attribute__((ext_vector_type(8)))  float    v8f;

// ---------------- packed f16 weight layout in d_ws (element offsets) -------
// Layer geometry: K padded to mult of 32, stored per-fragment so each lane's
// 16 B-matrix halves are contiguous (32B): idx = ((c*NT + t)*32 + lane)*16 + j
// lane<16 covers K = c*32 + 0..15 ; lane>=16 covers K = c*32 + 16..31 (j = K%16)
#define OFF_PY   0         // [160 x 256] W_prior_y (129 real rows)
#define OFF_QY   40960     // [256 x 256] W_q_y
#define OFF_PZ   106496    // [ 64 x 256] W_prior_z
#define OFF_PMS  122880    // [256 x 128] W_prior_z_mean || W_prior_z_std
#define OFF_ENC  155648    // [288 x 256] W_enc (257 real rows)      -> LDS
#define OFF_EMS  229376    // [256 x 128] W_enc_mean || W_enc_std    -> LDS
#define OFF_DEC  262144    // [224 x 256] W_dec (193 real rows)
#define OFF_DMS  319488    // [256 x 256] W_dec_mean || W_dec_std
#define OFF_RNN  385024    // [352 x 128] W_ih ; W_hh (321 real rows)-> LDS
#define WS_TOTAL 430080

// ---------------- LDS budget ----------------
#define LDS_ENC_H 73728
#define LDS_EMS_H 32768
#define LDS_RNN_H 45056
#define LDS_W_H   (LDS_ENC_H + LDS_EMS_H + LDS_RNN_H)   // 151552 halves
#define LDS_ACT_H 512
#define LDS_ST_H  512
#define SM_HALVES (LDS_W_H + LDS_ACT_H + LDS_ST_H)      // 152576 halves
#define SM_FLOATS (256 + 512 + 128 + 256 + 16 + 8)      // hid,vec,xc,red,scal,acc
#define SMEM_BYTES (SM_HALVES * 2 + SM_FLOATS * 4)      // 309,856 B (< 320KB WGP LDS)

#define CLOG (-0.9189385332046727f)  // -0.5*log(2*pi)

// =====================================================================
// prep: convert f32 weights -> f16 fragment-ordered layout in d_ws
// =====================================================================
__global__ __launch_bounds__(256) void vrnn_prep(
    const float* __restrict__ Wpy,  const float* __restrict__ Wqy,
    const float* __restrict__ Wpz,  const float* __restrict__ Wpzm,
    const float* __restrict__ Wpzs, const float* __restrict__ Wenc,
    const float* __restrict__ Wem,  const float* __restrict__ Wes,
    const float* __restrict__ Wdec, const float* __restrict__ Wdm,
    const float* __restrict__ Wds,  const float* __restrict__ Wih,
    const float* __restrict__ Whh,  _Float16* __restrict__ ws)
{
    int idx = blockIdx.x * blockDim.x + threadIdx.x;
    if (idx >= WS_TOTAL) return;
    const int bases[10] = {OFF_PY, OFF_QY, OFF_PZ, OFF_PMS, OFF_ENC,
                           OFF_EMS, OFF_DEC, OFF_DMS, OFF_RNN, WS_TOTAL};
    const int nts[9]    = {16, 16, 16, 8, 16, 8, 16, 16, 8};
    int L = 0;
    while (idx >= bases[L + 1]) ++L;
    int local  = idx - bases[L];
    int frag   = local >> 9;          // 512 halves per 32x16 fragment
    int within = local & 511;
    int lane   = within >> 4;
    int j      = within & 15;
    int NT     = nts[L];
    int t      = frag % NT;
    int c      = frag / NT;
    int k      = c * 32 + (lane & 16) + j;   // K index (B layout)
    int n      = t * 16 + (lane & 15);       // N index
    float v = 0.0f;
    switch (L) {
        case 0: v = (k < 129) ? Wpy[k * 256 + n] : 0.0f; break;
        case 1: v = Wqy[k * 256 + n]; break;
        case 2: v = Wpz[k * 256 + n]; break;
        case 3: v = (n < 64) ? Wpzm[k * 64 + n] : Wpzs[k * 64 + (n - 64)]; break;
        case 4: v = (k < 257) ? Wenc[k * 256 + n] : 0.0f; break;
        case 5: v = (n < 64) ? Wem[k * 64 + n] : Wes[k * 64 + (n - 64)]; break;
        case 6: v = (k < 193) ? Wdec[k * 256 + n] : 0.0f; break;
        case 7: v = (n < 128) ? Wdm[k * 128 + n] : Wds[k * 128 + (n - 128)]; break;
        case 8: v = (k < 193) ? Wih[k * 128 + n]
                              : ((k < 321) ? Whh[(k - 193) * 128 + n] : 0.0f); break;
    }
    ws[idx] = (_Float16)v;
}

// =====================================================================
// WMMA matvec: out[N] = act[K] @ W[K,N], A rows replicated so D row 0 = result.
// 8 waves split the N/16 tiles; each wave chains KC accumulating WMMAs.
// =====================================================================
template <int KC, int NT>
__device__ __forceinline__ void matvec_wmma(const _Float16* __restrict__ W,
                                            const _Float16* __restrict__ act,
                                            float* __restrict__ out,
                                            int lane, int wave)
{
    const int o = (lane & 16) ? 16 : 0;  // A-fragment K sub-offset per lane half
    for (int t = wave; t < NT; t += 8) {
        v8f acc = {};
#pragma unroll
        for (int c = 0; c < KC; ++c) {
            const int kb = c * 32;
            v8h lo = *(const v8h*)(act + kb + o);        // K = kb + o .. +7
            v8h hi = *(const v8h*)(act + kb + 8 + o);    // K = kb + 8 + o ..
            v16h a = __builtin_shufflevector(lo, hi, 0, 1, 2, 3, 4, 5, 6, 7,
                                             8, 9, 10, 11, 12, 13, 14, 15);
            v16h b = *(const v16h*)(W + ((c * NT + t) * 32 + lane) * 16);
            acc = __builtin_amdgcn_wmma_f32_16x16x32_f16(
                false, a, false, b, (short)0, acc, false, false);
        }
        if (lane < 16) out[t * 16 + lane] = acc[0];  // D VGPR0, lanes 0-15 = row M=0
    }
}

__device__ __forceinline__ float block_reduce(float v, float* sR, int tid)
{
#pragma unroll
    for (int off = 16; off > 0; off >>= 1) v += __shfl_down(v, off, 32);
    __syncthreads();                       // WAR guard on sR reuse
    if ((tid & 31) == 0) sR[tid >> 5] = v;
    __syncthreads();
    return sR[0] + sR[1] + sR[2] + sR[3] + sR[4] + sR[5] + sR[6] + sR[7];
}

__device__ __forceinline__ float softplusf(float v)
{
    return (v > 0.0f) ? (v + log1pf(expf(-v))) : log1pf(expf(v));
}
__device__ __forceinline__ float sigmoidf(float v) { return 1.0f / (1.0f + expf(-v)); }

// =====================================================================
// persistent single-workgroup scan kernel
// =====================================================================
__global__ __launch_bounds__(256, 1) void vrnn_run(
    const float* __restrict__ x,   const float* __restrict__ yv,
    const float* __restrict__ epz, const float* __restrict__ epu,
    const float* __restrict__ wpyp, const float* __restrict__ wqyp,
    const _Float16* __restrict__ ws, float* __restrict__ out)
{
    extern __shared__ _Float16 smem[];
    _Float16* sWenc = smem;                       // 73728 h
    _Float16* sWems = sWenc + LDS_ENC_H;          // 32768 h
    _Float16* sWrnn = sWems + LDS_EMS_H;          // 45056 h
    _Float16* sAct  = sWrnn + LDS_RNN_H;          // 512 h (per-layer activation)
    _Float16* sSt   = sAct + LDS_ACT_H;           // 512 h: [xc|h|z|xt] f16 state
    float* sHid  = (float*)(smem + SM_HALVES);    // 256 hidden f32
    float* sVec  = sHid + 256;                    // pms[0:128] ems[128:256] dms[256:512]
    float* sXc   = sVec + 512;                    // 128 (xc in f32 for rec loss)
    float* sRed  = sXc + 128;                     // 256 reduction scratch
    float* sScal = sRed + 256;                    // scalars ([2] = y carry)
    float* sAcc  = sScal + 16;                    // 7 loss accumulators

    const int tid  = threadIdx.x;
    const int lane = tid & 31;
    const int wave = tid >> 5;

    // ---- stage chain-critical weights into LDS (ENC+EMS contiguous, then RNN)
    {
        const uint4* s1 = (const uint4*)(ws + OFF_ENC);
        uint4* d1 = (uint4*)sWenc;
        for (int i = tid; i < (LDS_ENC_H + LDS_EMS_H) / 8; i += 256) d1[i] = s1[i];
        const uint4* s2 = (const uint4*)(ws + OFF_RNN);
        uint4* d2 = (uint4*)sWrnn;
        for (int i = tid; i < LDS_RNN_H / 8; i += 256) d2[i] = s2[i];
    }
    for (int i = tid; i < 512; i += 256) { sSt[i] = (_Float16)0.0f; sAct[i] = (_Float16)0.0f; }
    if (tid < 16) sScal[tid] = 0.0f;
    if (tid < 8)  sAcc[tid & 7] = 0.0f;
    __syncthreads();

    const _Float16* wsPY  = ws + OFF_PY;
    const _Float16* wsQY  = ws + OFF_QY;
    const _Float16* wsPZ  = ws + OFF_PZ;
    const _Float16* wsPMS = ws + OFF_PMS;
    const _Float16* wsDEC = ws + OFF_DEC;
    const _Float16* wsDMS = ws + OFF_DMS;

    for (int t = 0; t < T_STEPS; ++t) {
        if (t + 1 < T_STEPS) {   // hint next step's inputs toward the WGP
            __builtin_prefetch(x + (size_t)(t + 1) * XDIM, 0, 1);
            __builtin_prefetch(epz + (size_t)(t + 1) * ZDIM, 0, 1);
        }
        const float y_in = yv[t];
        const float lf   = (y_in != -1.0f) ? 1.0f : 0.0f;
        const float eu   = epu[t];
        const float y_prev = sScal[2];

        // xc (new) + act_py = [x_prev(128), y_prev, 0-pad to 160]
        if (tid < XDIM) {
            float xcf = x[(size_t)t * XDIM + tid];
            sXc[tid] = xcf;
            sSt[tid] = (_Float16)xcf;
            float xp = x[(size_t)((t == 0) ? (T_STEPS - 1) : (t - 1)) * XDIM + tid];
            sAct[tid] = (_Float16)xp;
        } else if (tid < 160) {
            sAct[tid] = (tid == 128) ? (_Float16)y_prev : (_Float16)0.0f;
        }
        __syncthreads();

        // p(y) : hidden [160->256] then scalar head
        matvec_wmma<5, 16>(wsPY, sAct, sHid, lane, wave);
        __syncthreads();
        float p_lin = block_reduce(fmaxf(sHid[tid], 0.0f) * wpyp[tid], sRed, tid);
        const float p_yt = sigmoidf(p_lin);

        // q(y) : act = [xc,h] = state[0:256] directly
        matvec_wmma<8, 16>(wsQY, sSt, sHid, lane, wave);
        __syncthreads();
        float q_lin = block_reduce(fmaxf(sHid[tid], 0.0f) * wqyp[tid], sRed, tid);
        const float q_yt = sigmoidf(q_lin);

        // gumbel-sigmoid relaxed y_t
        const float gum = sigmoidf(logf(eu) - logf(1.0f - eu) + logf(q_yt) - logf(1.0f - q_yt));
        const float y_t = lf * y_in + (1.0f - lf) * gum;
        if (tid == 0) sScal[2] = y_t;

        // prior z: pzh = relu(z @ Wpz) ; [pm|ps_raw] = pzh @ Wpms
        matvec_wmma<2, 16>(wsPZ, sSt + 256, sHid, lane, wave);
        __syncthreads();
        sAct[tid] = (_Float16)fmaxf(sHid[tid], 0.0f);
        __syncthreads();
        matvec_wmma<8, 8>(wsPMS, sAct, sVec, lane, wave);
        __syncthreads();

        // encoder: act = [xc, y_t, h, 0-pad to 288] (LDS weights)
        for (int i = tid; i < 288; i += 256) {
            _Float16 v;
            if (i < 128)       v = sSt[i];
            else if (i == 128) v = (_Float16)y_t;
            else if (i < 257)  v = sSt[128 + (i - 129)];
            else               v = (_Float16)0.0f;
            sAct[i] = v;
        }
        __syncthreads();
        matvec_wmma<9, 16>(sWenc, sAct, sHid, lane, wave);
        __syncthreads();
        sAct[tid] = (_Float16)fmaxf(sHid[tid], 0.0f);
        __syncthreads();
        matvec_wmma<8, 8>(sWems, sAct, sVec + 128, lane, wave);
        __syncthreads();

        // z_t = ez*es + em ; per-element KLD vs prior
        float kv = 0.0f;
        if (tid < ZDIM) {
            float em = sVec[128 + tid], es = softplusf(sVec[192 + tid]);
            float pm = sVec[tid],       ps = softplusf(sVec[64 + tid]);
            float ez = epz[(size_t)t * ZDIM + tid];
            sSt[256 + tid] = (_Float16)(ez * es + em);
            float d = em - pm;
            kv = logf(ps / es) + (es * es + d * d) / (2.0f * ps * ps) - 0.5f;
        }
        const float kld = block_reduce(kv, sRed, tid);

        // decoder: act = [x_prev_carry(128), z_t(64), y_t, 0-pad to 224]
        if (tid < 224) {
            _Float16 v;
            if (tid < 128)      v = sSt[320 + tid];
            else if (tid < 192) v = sSt[256 + (tid - 128)];
            else if (tid == 192) v = (_Float16)y_t;
            else                v = (_Float16)0.0f;
            sAct[tid] = v;
        }
        __syncthreads();
        matvec_wmma<7, 16>(wsDEC, sAct, sHid, lane, wave);
        __syncthreads();
        sAct[tid] = (_Float16)fmaxf(sHid[tid], 0.0f);
        __syncthreads();
        matvec_wmma<8, 16>(wsDMS, sAct, sVec + 256, lane, wave);
        __syncthreads();

        // reconstruction loss
        float rv = 0.0f;
        if (tid < XDIM) {
            float dm = sVec[256 + tid], ds = softplusf(sVec[384 + tid]);
            float diff = sXc[tid] - dm;
            rv = CLOG + logf(ds) + diff * diff / (2.0f * ds * ds);
        }
        const float rec = block_reduce(rv, sRed, tid);

        // RNN: act = [y_t, z_t(64), xc(128), h(128), 0-pad to 352] (LDS weights)
        for (int i = tid; i < 352; i += 256) {
            _Float16 v;
            if (i == 0)        v = (_Float16)y_t;
            else if (i < 65)   v = sSt[256 + (i - 1)];
            else if (i < 193)  v = sSt[i - 65];
            else if (i < 321)  v = sSt[128 + (i - 193)];
            else               v = (_Float16)0.0f;
            sAct[i] = v;
        }
        __syncthreads();
        matvec_wmma<11, 8>(sWrnn, sAct, sHid, lane, wave);
        __syncthreads();
        if (tid < HDIM) {
            sSt[128 + tid] = (_Float16)tanhf(sHid[tid]);  // h_new
            sSt[320 + tid] = sSt[tid];                    // x carry <- xc
        }
        if (tid == 0) {
            float p = p_yt, q = q_yt, ul = 1.0f - lf;
            float bce  = -(y_t * logf(p) + (1.0f - y_t) * logf(1.0f - p));
            float addt =  y_t * logf(p * q) + (1.0f - y_t) * logf((1.0f - p) * (1.0f - q));
            float kcat =  p * logf(p / q) + (1.0f - p) * logf((1.0f - p) / (1.0f - q));
            sAcc[0] += lf * kld;  sAcc[1] += lf * rec;  sAcc[2] += lf * bce;
            sAcc[3] += ul * kld;  sAcc[4] += ul * rec;  sAcc[5] += ul * kcat;
            sAcc[6] += lf * addt;
        }
        __syncthreads();
    }

    if (tid == 0)
        for (int i = 0; i < 7; ++i) out[i] = sAcc[i];
}

// =====================================================================
extern "C" void kernel_launch(void* const* d_in, const int* in_sizes, int n_in,
                              void* d_out, int out_size, void* d_ws, size_t ws_size,
                              hipStream_t stream)
{
    (void)in_sizes; (void)n_in; (void)out_size; (void)ws_size;
    const float* x    = (const float*)d_in[0];
    const float* y    = (const float*)d_in[1];
    const float* epz  = (const float*)d_in[2];
    const float* epu  = (const float*)d_in[3];
    const float* Wpz  = (const float*)d_in[4];
    const float* Wpzm = (const float*)d_in[5];
    const float* Wpzs = (const float*)d_in[6];
    const float* Wpy  = (const float*)d_in[7];
    const float* wpyp = (const float*)d_in[8];
    const float* Wqy  = (const float*)d_in[9];
    const float* wqyp = (const float*)d_in[10];
    const float* Wenc = (const float*)d_in[11];
    const float* Wem  = (const float*)d_in[12];
    const float* Wes  = (const float*)d_in[13];
    const float* Wdec = (const float*)d_in[14];
    const float* Wdm  = (const float*)d_in[15];
    const float* Wds  = (const float*)d_in[16];
    const float* Wih  = (const float*)d_in[17];
    const float* Whh  = (const float*)d_in[18];
    _Float16* ws = (_Float16*)d_ws;

    vrnn_prep<<<(WS_TOTAL + 255) / 256, 256, 0, stream>>>(
        Wpy, Wqy, Wpz, Wpzm, Wpzs, Wenc, Wem, Wes, Wdec, Wdm, Wds, Wih, Whh, ws);

    (void)hipFuncSetAttribute(reinterpret_cast<const void*>(vrnn_run),
                              hipFuncAttributeMaxDynamicSharedMemorySize,
                              (int)SMEM_BYTES);
    vrnn_run<<<1, 256, SMEM_BYTES, stream>>>(x, y, epz, epu, wpyp, wqyp, ws,
                                             (float*)d_out);
}
